// Mahalanobis_72834055405642
// MI455X (gfx1250) — matrix operand order repaired and checked by reading, not compiled, not run on hardware
//
#include <hip/hip_runtime.h>

#define B_  256
#define N_  512
#define C_  1000
#define R_  64
#define KT  32

typedef __attribute__((ext_vector_type(16))) __bf16 v16bf;
typedef __attribute__((ext_vector_type(8)))  __bf16 v8bf;
typedef __attribute__((ext_vector_type(8)))  float  v8f;
typedef __attribute__((ext_vector_type(4)))  unsigned int u32x4;
typedef __attribute__((ext_vector_type(8)))  int    i32x8;
typedef __attribute__((ext_vector_type(4)))  int    i32x4;

#if defined(__gfx1250__) && __has_builtin(__builtin_amdgcn_tensor_load_to_lds)
#define USE_TDM 1
#else
#define USE_TDM 0
#endif

__global__ __launch_bounds__(256, 2)
void mahalanobis_wmma_kernel(const float* __restrict__ x,
                             const float* __restrict__ weight,
                             const float* __restrict__ bias,
                             float* __restrict__ out)
{
    // LDS staging
    __shared__ __bf16 xs[B_][KT + 8];    // A tile (b-major), padded rows (80B)
    __shared__ __bf16 ws[R_][KT + 16];   // B tile [r][k], padded rows (96B)
    __shared__ float  wtile[R_ * KT];    // f32 W tile (TDM destination)
    __shared__ float  wbpart[256];
    __shared__ float  wb[R_];

    const int tid  = threadIdx.x;
    const int lane = tid & 31;
    const int wave = tid >> 5;
    const int l15  = lane & 15;
    const int half = lane >> 4;          // 0: lanes 0-15, 1: lanes 16-31
    const int c    = blockIdx.x;

    const v8f vzero = {0.f,0.f,0.f,0.f,0.f,0.f,0.f,0.f};
    v8f acc[2][4];
#pragma unroll
    for (int mi = 0; mi < 2; ++mi)
#pragma unroll
        for (int ni = 0; ni < 4; ++ni)
            acc[mi][ni] = vzero;

    float wbacc = 0.f;

    const int wr = tid >> 2;             // 0..63 : row of W tile this thread converts
    const int wk = (tid & 3) * 8;        // 0,8,16,24 : k sub-chunk

#if USE_TDM
    const unsigned ldsW = (unsigned)(uintptr_t)(void*)wtile;
#endif

    for (int k0 = 0; k0 < N_; k0 += KT) {
        __syncthreads();                 // previous iteration's LDS consumers done

#if USE_TDM
        if (wave == 0) {
            unsigned long long ga =
                (unsigned long long)(uintptr_t)(weight + ((size_t)c * R_ * N_ + k0));
            u32x4 g0;
            g0[0] = 1u;                                            // count=1, user desc
            g0[1] = ldsW;                                          // lds_addr
            g0[2] = (unsigned)(ga & 0xFFFFFFFFu);                  // global_addr[31:0]
            g0[3] = (unsigned)((ga >> 32) & 0x1FFFFFFu) | (2u << 30); // addr[56:32] | type=2
            i32x8 g1;
            g1[0] = (int)(2u << 16);                               // data_size=4B
            g1[1] = (int)((unsigned)N_ << 16);                     // tensor_dim0 = 512
            g1[2] = (int)((unsigned)R_ << 16);                     // tensor_dim1 = 64
            g1[3] = (int)((unsigned)KT << 16);                     // tile_dim0 = 32
            g1[4] = (int)R_;                                       // tile_dim1 = 64, tile_dim2 = 0
            g1[5] = (int)N_;                                       // tensor_dim0_stride = 512
            g1[6] = 0;
            g1[7] = 0;
            i32x4 gz4 = {0, 0, 0, 0};
            i32x8 gz8 = {0, 0, 0, 0, 0, 0, 0, 0};
            __builtin_amdgcn_tensor_load_to_lds(g0, g1, gz4, gz4, gz8, 0);
        }
#else
        {
            const float* wp = weight + (size_t)c * R_ * N_ + (size_t)wr * N_ + k0 + wk;
            float4 a = *(const float4*)(wp + 0);
            float4 b = *(const float4*)(wp + 4);
            float* dst = &wtile[wr * KT + wk];
            *(float4*)(dst + 0) = a;
            *(float4*)(dst + 4) = b;
        }
#endif

        // Stage x tile as bf16: thread t owns batch row t (overlaps with TDM)
        {
            const float* xp = x + (size_t)tid * N_ + k0;
#pragma unroll
            for (int j = 0; j < KT; j += 4) {
                float4 v = *(const float4*)(xp + j);
                xs[tid][j + 0] = (__bf16)v.x;
                xs[tid][j + 1] = (__bf16)v.y;
                xs[tid][j + 2] = (__bf16)v.z;
                xs[tid][j + 3] = (__bf16)v.w;
            }
        }

#if USE_TDM
        if (wave == 0) {
            __builtin_amdgcn_s_wait_tensorcnt(0);
        }
#endif
        __syncthreads();                 // wtile + xs visible to all waves

        // Convert W tile to bf16 (once) and accumulate Wb partial in f32
        {
            const float* bp = bias + (size_t)c * N_ + k0 + wk;
#pragma unroll
            for (int j = 0; j < 8; ++j) {
                float wv = wtile[wr * KT + wk + j];
                ws[wr][wk + j] = (__bf16)wv;
                wbacc += wv * bp[j];
            }
        }
        __syncthreads();                 // ws ready

        // Fragment assembly per ISA 16-bit layouts
        const int kofA = half * 8;       // A: lanes 0-15 K=[0,8)+[16,24); 16-31 K=[8,16)+[24,32)
        const int kofB = half * 8;       // B: lanes 0-15 K=[0,8)+[16,24); 16-31 K=[8,16)+[24,32)

        v16bf afrag[2];
#pragma unroll
        for (int mi = 0; mi < 2; ++mi) {
            const int row = wave * 32 + mi * 16 + l15;
            v8bf lo = *(const v8bf*)&xs[row][kofA];
            v8bf hi = *(const v8bf*)&xs[row][kofA + 16];
            afrag[mi] = __builtin_shufflevector(lo, hi,
                0,1,2,3,4,5,6,7,8,9,10,11,12,13,14,15);
        }
        v16bf bfrag[4];
#pragma unroll
        for (int ni = 0; ni < 4; ++ni) {
            const __bf16* p = &ws[ni * 16 + l15][kofB];
            v8bf lo = *(const v8bf*)(p + 0);
            v8bf hi = *(const v8bf*)(p + 16);
            bfrag[ni] = __builtin_shufflevector(lo, hi,
                0,1,2,3,4,5,6,7,8,9,10,11,12,13,14,15);
        }

#pragma unroll
        for (int mi = 0; mi < 2; ++mi)
#pragma unroll
            for (int ni = 0; ni < 4; ++ni)
                acc[mi][ni] = __builtin_amdgcn_wmma_f32_16x16x32_bf16(
                    /*neg_a=*/false, afrag[mi],
                    /*neg_b=*/false, bfrag[ni],
                    /*c_mod=*/(short)0, acc[mi][ni],
                    /*reuse_a=*/false, /*reuse_b=*/false);
    }

    // Reduce Wb partials: wb[r] = sum of 4 partials
    wbpart[tid] = wbacc;
    __syncthreads();
    if (tid < R_)
        wb[tid] = wbpart[4 * tid] + wbpart[4 * tid + 1] +
                  wbpart[4 * tid + 2] + wbpart[4 * tid + 3];
    __syncthreads();

    // Epilogue: out[b,c] = sum_r (s[b,r] - wb[r])^2
    float wbr[4];
#pragma unroll
    for (int ni = 0; ni < 4; ++ni) wbr[ni] = wb[ni * 16 + l15];

#pragma unroll
    for (int mi = 0; mi < 2; ++mi) {
#pragma unroll
        for (int j = 0; j < 8; ++j) {
            float p = 0.f;
#pragma unroll
            for (int ni = 0; ni < 4; ++ni) {
                float e = acc[mi][ni][j] - wbr[ni];
                p += e * e;
            }
            // horizontal sum across 16 lanes (each half of wave32 independently)
            p += __shfl_xor(p, 1, 32);
            p += __shfl_xor(p, 2, 32);
            p += __shfl_xor(p, 4, 32);
            p += __shfl_xor(p, 8, 32);
            if (l15 == 0) {
                const int row = wave * 32 + mi * 16 + j + half * 8;
                out[(size_t)row * C_ + c] = p;
            }
        }
    }
}

extern "C" void kernel_launch(void* const* d_in, const int* in_sizes, int n_in,
                              void* d_out, int out_size, void* d_ws, size_t ws_size,
                              hipStream_t stream) {
    (void)in_sizes; (void)n_in; (void)d_ws; (void)ws_size; (void)out_size;
    const float* x      = (const float*)d_in[0];
    const float* weight = (const float*)d_in[1];
    const float* bias   = (const float*)d_in[2];
    float* out          = (float*)d_out;
    dim3 grid(C_), block(256);
    hipLaunchKernelGGL(mahalanobis_wmma_kernel, grid, block, 0, stream,
                       x, weight, bias, out);
}
